// LSTMCRFModel_7035156431054
// MI455X (gfx1250) — matrix operand order repaired and checked by reading, not compiled, not run on hardware
//
#include <hip/hip_runtime.h>
#include <hip/hip_bf16.h>
#include <stdint.h>
#include <stddef.h>

typedef __bf16 bf16;
typedef __attribute__((ext_vector_type(16))) __bf16 v16bf;
typedef __attribute__((ext_vector_type(8)))  __bf16 bf16x8;
typedef __attribute__((ext_vector_type(4)))  __bf16 bf16x4;
typedef __attribute__((ext_vector_type(8)))  float  v8f;
typedef __attribute__((ext_vector_type(4)))  float  f32x4;

#define N_B   64
#define N_T   512
#define N_BT  32768     // B*T
#define N_E   256
#define N_H   256
#define N_2H  512
#define N_G   1024      // 4*H gate width
#define N_K   32        // CRF tags

// ---------------- device helpers ----------------

__device__ __forceinline__ float sigm_f(float x) { return 1.f / (1.f + __expf(-x)); }
__device__ __forceinline__ float tanh_f(float x) {
  float e = __expf(2.f * x);
  return (e - 1.f) / (e + 1.f);
}

// A fragment (16x32 bf16, MxK): lane holds row m=lane&15, K = k0+klane+{0..15} contiguous,
// klane = (lane>>4)*16.  Two 16B loads -> VGPR halves.
__device__ __forceinline__ v16bf ldfragA(const bf16* p) {
  union { v16bf v; bf16x8 h[2]; } u;
  u.h[0] = *(const bf16x8*)(p);
  u.h[1] = *(const bf16x8*)(p + 8);
  return u.v;
}
// B fragment (32x16 bf16, KxN) from row-major [N,K] weights: lane holds col n=lane&15,
// K = k0 + (lane>>4)*16 + {0..15} contiguous.
__device__ __forceinline__ v16bf ldfragB(const bf16* p) {
  union { v16bf v; bf16x8 h[2]; } u;
  u.h[0] = *(const bf16x8*)(p);
  u.h[1] = *(const bf16x8*)(p + 8);
  return u.v;
}

// ---------------- elementwise kernels ----------------

__global__ __launch_bounds__(256) void cast_f32_bf16(const float* __restrict__ a,
                                                     bf16* __restrict__ o, int n) {
  int i = blockIdx.x * 256 + threadIdx.x;
  if (i < n) o[i] = (bf16)a[i];
}

// embedding gather: x0[bt,e] = bf16(emb_table[ids[bt], e]), 4 elems/thread
__global__ __launch_bounds__(256) void embed_bf16(const int* __restrict__ ids,
                                                  const float* __restrict__ table,
                                                  bf16* __restrict__ x) {
  size_t i = ((size_t)blockIdx.x * 256 + threadIdx.x) * 4;
  if (i >= (size_t)N_BT * N_E) return;
  int bt = (int)(i >> 8);          // / 256
  int e  = (int)(i & 255);
  const float* src = table + (size_t)ids[bt] * N_E + e;
  f32x4 v = *(const f32x4*)src;
  *(bf16x4*)(x + i) = __builtin_convertvector(v, bf16x4);
}

// ---------------- WMMA GEMM: C[M,NN]f32 = A[M,KC]bf16 * B[NN,KC]bf16^T + bias[NN] ------
// Shapes are compile-time so accumulators are statically indexed (no v_movrel) and the
// per-K-step WMMAs on independent accumulators hide the bf16 WMMA->WMMA RAW hazard.
// One wave computes a 16(M) x 64(N) strip (NSUB subtiles). M multiple of 16.

template <int NN, int KC>
__global__ __launch_bounds__(256)
void gemm_bf16_wmma(const bf16* __restrict__ A, const bf16* __restrict__ Bw,
                    const float* __restrict__ bias, float* __restrict__ C, int M) {
  constexpr int NSUB   = (NN >= 64) ? 4 : (NN / 16);
  constexpr int STRIPS = (NN + 63) / 64;

  const int lane = threadIdx.x & 31;
  int gwave = (int)((blockIdx.x * 256 + threadIdx.x) >> 5);
  gwave = __builtin_amdgcn_readfirstlane(gwave);
  const int mt = gwave / STRIPS;
  const int ns = gwave % STRIPS;
  if (mt * 16 >= M) return;

  const int lm = lane & 15;
  const int kA = (lane >> 4) * 16;   // A frag K sub-offset
  const int kB = (lane >> 4) * 16;   // B frag K sub-offset
  const int nb = ns * 64;

  v8f acc[NSUB];
#pragma unroll
  for (int s = 0; s < NSUB; ++s) {
    float bv = bias[nb + s * 16 + lm];
#pragma unroll
    for (int r = 0; r < 8; ++r) acc[s][r] = bv;
  }

  const bf16* Arow = A + (size_t)(mt * 16 + lm) * KC;
  const bf16* Brow = Bw + (size_t)(nb + lm) * KC;
#pragma unroll 4
  for (int k0 = 0; k0 < KC; k0 += 32) {
    v16bf af = ldfragA(Arow + k0 + kA);
    if (k0 + 32 < KC) __builtin_prefetch(Arow + k0 + 32 + kA, 0, 1);  // global_prefetch_b8
#pragma unroll
    for (int s = 0; s < NSUB; ++s) {
      v16bf bf_ = ldfragB(Brow + (size_t)(s * 16) * KC + k0 + kB);
      acc[s] = __builtin_amdgcn_wmma_f32_16x16x32_bf16(
          false, af, false, bf_, (short)0, acc[s], false, false);
    }
  }

  const int mbase = mt * 16 + ((lane >= 16) ? 8 : 0);
#pragma unroll
  for (int s = 0; s < NSUB; ++s) {
    int n = nb + s * 16 + lm;
#pragma unroll
    for (int r = 0; r < 8; ++r)
      C[(size_t)(mbase + r) * NN + n] = acc[s][r];
  }
}

// ---------------- recurrent LSTM scan (one direction-layer) ----------------
// Single 1024-thread workgroup (32 waves). Per step: hg[64,1024] = h[64,256] @ w_hh^T
// via WMMA (accumulators seeded from precomputed xg), in-register LSTM cell update,
// h ping-pongs through double-buffered LDS; c state lives in VGPRs across all 512 steps.
// slot = wave*2+s in [0,64): mt = slot>>4 (batch tile), j = slot&15 (hidden-col tile);
// the 4 gates (N-tiles j, j+16, j+32, j+48) belong to the same wave.

__global__ __launch_bounds__(1024)
void lstm_scan(const float* __restrict__ xg,   // [B,T,4H] f32 gate preacts
               const bf16* __restrict__ whh,   // [4H,H] bf16
               bf16* __restrict__ out,         // [B,T,2H] bf16
               int reverse, int outOff) {
  __shared__ bf16 hbuf[2][N_B][N_H];           // 64 KB double buffer
  const int tid  = threadIdx.x;
  const int lane = tid & 31;
  const int wave = __builtin_amdgcn_readfirstlane(tid >> 5);
  const int lm   = lane & 15;
  const int hi8  = (lane >> 4) * 8;
  const int kB   = (lane >> 4) * 16;

  for (int i = tid; i < 2 * N_B * N_H; i += 1024) (&hbuf[0][0][0])[i] = (bf16)0.f;
  __syncthreads();

  v8f cst[2];
#pragma unroll
  for (int s = 0; s < 2; ++s) {
#pragma unroll
    for (int r = 0; r < 8; ++r) cst[s][r] = 0.f;
  }

  for (int t = 0; t < N_T; ++t) {
    const int te = reverse ? (N_T - 1 - t) : t;
    const bf16* hcur = &hbuf[t & 1][0][0];
    bf16* hnxt = &hbuf[(t + 1) & 1][0][0];

    v8f acc[2][4];
#pragma unroll
    for (int s = 0; s < 2; ++s) {
      const int slot = wave * 2 + s;
      const int mt = slot >> 4, j = slot & 15;
#pragma unroll
      for (int g = 0; g < 4; ++g) {
#pragma unroll
        for (int r = 0; r < 8; ++r) {
          int b = mt * 16 + r + hi8;
          acc[s][g][r] = xg[((size_t)b * N_T + te) * N_G + g * N_H + j * 16 + lm];
        }
      }
    }

#pragma unroll 2
    for (int k0 = 0; k0 < N_H; k0 += 32) {
#pragma unroll
      for (int s = 0; s < 2; ++s) {
        const int slot = wave * 2 + s;
        const int mt = slot >> 4, j = slot & 15;
        v16bf af = ldfragA(hcur + (size_t)(mt * 16 + lm) * N_H + k0 + kB);
#pragma unroll
        for (int g = 0; g < 4; ++g) {
          int n = g * N_H + j * 16 + lm;
          v16bf bf_ = ldfragB(whh + (size_t)n * N_H + k0 + kB);
          acc[s][g] = __builtin_amdgcn_wmma_f32_16x16x32_bf16(
              false, af, false, bf_, (short)0, acc[s][g], false, false);
        }
      }
    }

#pragma unroll
    for (int s = 0; s < 2; ++s) {
      const int slot = wave * 2 + s;
      const int mt = slot >> 4, j = slot & 15;
      const int col = j * 16 + lm;
#pragma unroll
      for (int r = 0; r < 8; ++r) {
        int b = mt * 16 + r + hi8;
        float gi = acc[s][0][r], gf = acc[s][1][r];
        float gg = acc[s][2][r], go = acc[s][3][r];
        float c = sigm_f(gf) * cst[s][r] + sigm_f(gi) * tanh_f(gg);
        cst[s][r] = c;
        float h = sigm_f(go) * tanh_f(c);
        bf16 hb = (bf16)h;
        hnxt[(size_t)b * N_H + col] = hb;
        out[((size_t)b * N_T + te) * N_2H + outOff + col] = hb;
      }
    }
    __syncthreads();
  }
}

// ---------------- CRF loss: lane = tag (K=32 == wave32) ----------------

__global__ __launch_bounds__(1024)
void crf_loss_kernel(const float* __restrict__ em, const int* __restrict__ labels,
                     const float* __restrict__ st, const float* __restrict__ en,
                     const float* __restrict__ tr, float* __restrict__ loss_out) {
  __shared__ float red[32];
  const int tid = threadIdx.x, lane = tid & 31;
  const int wave = __builtin_amdgcn_readfirstlane(tid >> 5);

  float trc[N_K];                 // column tr[k][lane] for all k
#pragma unroll
  for (int k = 0; k < N_K; ++k) trc[k] = tr[k * N_K + lane];

  float total = 0.f;
  for (int bi = 0; bi < 2; ++bi) {
    const int b = wave * 2 + bi;
    const float* emp = em + (size_t)b * N_T * N_K;
    const int* lab = labels + b * N_T;

    // numerator (lane-parallel over t)
    float np = 0.f;
    for (int t = lane; t < N_T; t += 32) {
      np += emp[t * N_K + lab[t]];
      if (t < N_T - 1) np += tr[lab[t] * N_K + lab[t + 1]];
    }
    for (int off = 16; off; off >>= 1) np += __shfl_down(np, off, 32);
    np = __shfl(np, 0, 32);
    float num = np + st[lab[0]] + en[lab[N_T - 1]];

    // alpha logsumexp scan
    float alpha = st[lane] + emp[lane];
    for (int t = 1; t < N_T; ++t) {
      float m = -3.0e38f;
#pragma unroll 8
      for (int k = 0; k < N_K; ++k)
        m = fmaxf(m, __shfl(alpha, k, 32) + trc[k]);
      float ss = 0.f;
#pragma unroll 8
      for (int k = 0; k < N_K; ++k)
        ss += __expf(__shfl(alpha, k, 32) + trc[k] - m);
      alpha = m + __logf(ss) + emp[t * N_K + lane];
    }
    // logZ over lanes
    float v = alpha + en[lane];
    float m = v;
    for (int off = 16; off; off >>= 1) m = fmaxf(m, __shfl_xor(m, off, 32));
    float ss = __expf(v - m);
    for (int off = 16; off; off >>= 1) ss += __shfl_xor(ss, off, 32);
    float logZ = m + __logf(ss);
    if (lane == 0) total += -(num - logZ);
  }
  if (lane == 0) red[wave] = total;
  __syncthreads();
  if (tid == 0) {
    float s = 0.f;
    for (int w = 0; w < 32; ++w) s += red[w];
    loss_out[0] = s;
  }
}

// ---------------- Viterbi: one wave per sequence ----------------

__global__ __launch_bounds__(256)
void viterbi_kernel(const float* __restrict__ em, const float* __restrict__ st,
                    const float* __restrict__ en, const float* __restrict__ tr,
                    int* __restrict__ bp, float* __restrict__ preds) {
  const int lane = threadIdx.x & 31;
  const int b = __builtin_amdgcn_readfirstlane((int)((blockIdx.x * 256 + threadIdx.x) >> 5));
  if (b >= N_B) return;
  const float* emp = em + (size_t)b * N_T * N_K;
  int* bpb = bp + (size_t)b * N_T * N_K;

  float trc[N_K];
#pragma unroll
  for (int k = 0; k < N_K; ++k) trc[k] = tr[k * N_K + lane];

  float score = st[lane] + emp[lane];
  for (int t = 1; t < N_T; ++t) {
    float best = -3.0e38f; int bix = 0;
#pragma unroll 8
    for (int k = 0; k < N_K; ++k) {
      float v = __shfl(score, k, 32) + trc[k];
      if (v > best) { best = v; bix = k; }   // strict '>' keeps first max (argmax)
    }
    bpb[t * N_K + lane] = bix;
    score = best + emp[t * N_K + lane];
  }
  // argmax over lanes of score + end (ties -> lowest index)
  float v = score + en[lane]; int idx = lane;
  for (int off = 16; off; off >>= 1) {
    float ov = __shfl_xor(v, off, 32);
    int   oi = __shfl_xor(idx, off, 32);
    if (ov > v || (ov == v && oi < idx)) { v = ov; idx = oi; }
  }
  // ensure sibling-lane bp stores are globally visible before lane-0 backtrack
  asm volatile("s_wait_storecnt 0x0" ::: "memory");
  if (lane == 0) {
    int cur = idx;
    preds[(size_t)b * N_T + (N_T - 1)] = (float)cur;
    for (int t = N_T - 1; t >= 1; --t) {
      cur = bpb[t * N_K + cur];
      preds[(size_t)b * N_T + (t - 1)] = (float)cur;
    }
  }
}

// ---------------- host ----------------

static inline size_t align256(size_t x) { return (x + 255) & ~(size_t)255; }

extern "C" void kernel_launch(void* const* d_in, const int* in_sizes, int n_in,
                              void* d_out, int out_size, void* d_ws, size_t ws_size,
                              hipStream_t stream) {
  (void)in_sizes; (void)n_in; (void)out_size; (void)ws_size;

  const int*   ids     = (const int*)  d_in[0];
  const int*   labels  = (const int*)  d_in[1];
  // d_in[2] = mask (all ones; reference ignores it)
  const float* emb_tab = (const float*)d_in[3];
  const float* w_ih[4] = { (const float*)d_in[4],  (const float*)d_in[7],
                           (const float*)d_in[10], (const float*)d_in[13] }; // 0f,0b,1f,1b
  const float* w_hh[4] = { (const float*)d_in[5],  (const float*)d_in[8],
                           (const float*)d_in[11], (const float*)d_in[14] };
  const float* bias[4] = { (const float*)d_in[6],  (const float*)d_in[9],
                           (const float*)d_in[12], (const float*)d_in[15] };
  const float* w_out   = (const float*)d_in[16];
  const float* b_out   = (const float*)d_in[17];
  const float* crf_st  = (const float*)d_in[18];
  const float* crf_en  = (const float*)d_in[19];
  const float* crf_tr  = (const float*)d_in[20];
  float* out = (float*)d_out;

  // workspace layout
  char* w = (char*)d_ws;
  size_t off = 0;
  auto take = [&](size_t bytes) { char* p = w + off; off += align256(bytes); return p; };
  bf16*  x0      = (bf16*) take((size_t)N_BT * N_E * 2);
  float* xg_f    = (float*)take((size_t)N_BT * N_G * 4);    // reused layer0->layer1
  float* xg_b    = (float*)take((size_t)N_BT * N_G * 4);
  bf16*  out0    = (bf16*) take((size_t)N_BT * N_2H * 2);
  bf16*  out1    = (bf16*) take((size_t)N_BT * N_2H * 2);
  float* em      = (float*)take((size_t)N_BT * N_K * 4);
  int*   bp      = (int*)  take((size_t)N_B * N_T * N_K * 4);
  bf16*  wih_bf[4]; bf16* whh_bf[4];
  const int wih_n[4] = { N_G * N_E, N_G * N_E, N_G * N_2H, N_G * N_2H };
  for (int i = 0; i < 4; ++i) wih_bf[i] = (bf16*)take((size_t)wih_n[i] * 2);
  for (int i = 0; i < 4; ++i) whh_bf[i] = (bf16*)take((size_t)N_G * N_H * 2);
  bf16* wout_bf  = (bf16*) take((size_t)N_K * N_2H * 2);

  // 1) weight casts to bf16
  for (int i = 0; i < 4; ++i) {
    cast_f32_bf16<<<(wih_n[i] + 255) / 256, 256, 0, stream>>>(w_ih[i], wih_bf[i], wih_n[i]);
    cast_f32_bf16<<<(N_G * N_H + 255) / 256, 256, 0, stream>>>(w_hh[i], whh_bf[i], N_G * N_H);
  }
  cast_f32_bf16<<<(N_K * N_2H + 255) / 256, 256, 0, stream>>>(w_out, wout_bf, N_K * N_2H);

  // 2) embedding gather -> bf16
  embed_bf16<<<(N_BT * N_E / 4 + 255) / 256, 256, 0, stream>>>(ids, emb_tab, x0);

  auto gemm_blocks = [](int M, int N) {
    int waves = (M / 16) * ((N + 63) >> 6);
    return (waves + 7) / 8;            // 8 waves per 256-thread block
  };

  // 3) layer 0: input-gate GEMMs then scans
  gemm_bf16_wmma<N_G, N_E><<<gemm_blocks(N_BT, N_G), 256, 0, stream>>>(x0, wih_bf[0], bias[0], xg_f, N_BT);
  gemm_bf16_wmma<N_G, N_E><<<gemm_blocks(N_BT, N_G), 256, 0, stream>>>(x0, wih_bf[1], bias[1], xg_b, N_BT);
  lstm_scan<<<1, 1024, 0, stream>>>(xg_f, whh_bf[0], out0, /*reverse=*/0, /*outOff=*/0);
  lstm_scan<<<1, 1024, 0, stream>>>(xg_b, whh_bf[1], out0, /*reverse=*/1, /*outOff=*/N_H);

  // 4) layer 1
  gemm_bf16_wmma<N_G, N_2H><<<gemm_blocks(N_BT, N_G), 256, 0, stream>>>(out0, wih_bf[2], bias[2], xg_f, N_BT);
  gemm_bf16_wmma<N_G, N_2H><<<gemm_blocks(N_BT, N_G), 256, 0, stream>>>(out0, wih_bf[3], bias[3], xg_b, N_BT);
  lstm_scan<<<1, 1024, 0, stream>>>(xg_f, whh_bf[2], out1, 0, 0);
  lstm_scan<<<1, 1024, 0, stream>>>(xg_b, whh_bf[3], out1, 1, N_H);

  // 5) emissions
  gemm_bf16_wmma<N_K, N_2H><<<gemm_blocks(N_BT, N_K), 256, 0, stream>>>(out1, wout_bf, b_out, em, N_BT);

  // 6) CRF loss -> out[0]
  crf_loss_kernel<<<1, 1024, 0, stream>>>(em, labels, crf_st, crf_en, crf_tr, out);

  // 7) Viterbi -> out[1..B*T]
  viterbi_kernel<<<(N_B * 32 + 255) / 256, 256, 0, stream>>>(em, crf_st, crf_en, crf_tr, bp, out + 1);
}
